// Logits_5970004541786
// MI455X (gfx1250) — compile-verified, matrix operand order checked by reading, not run on hardware
//
#include <hip/hip_runtime.h>
#include <math.h>

// Problem constants (from reference): B=128 batch, K=2048 classes, D=512 feat, c=1
#define B_SZ 128
#define K_SZ 2048
#define D_SZ 512
#define CURV 1.0f
#define NCHUNK (D_SZ / 32)   // 16 K-chunks of 32

typedef __attribute__((ext_vector_type(16))) __bf16 v16bf;
typedef __attribute__((ext_vector_type(8)))  __bf16 v8bf;
typedef __attribute__((ext_vector_type(8)))  float  v8f;

// ---------------------------------------------------------------------------
// Fragment-order column permutation for the 16x16x32 bf16 WMMA (ISA 7.12.2):
// within a 32-col chunk, lanes 0-15 consume K = {0..7,16..23}, lanes 16-31
// consume K = {8..15,24..31}.  Storing the chunk in order
//   [0..7, 16..23, 8..15, 24..31]
// makes each lane's 16 bf16 fragment a single contiguous 32-byte load.
// ---------------------------------------------------------------------------
__device__ __forceinline__ int frag_perm(int pos) {
  const int grp = pos >> 3;                 // 0..3
  return (grp == 1) ? pos + 8 : (grp == 2) ? pos - 8 : pos;
}

// ---------------------------------------------------------------------------
// Pre-kernel 1: per-row scalars.  One wave per row (wave32).
//   rows [0,K):  uu[k] = ||p_k||^2,  ua[k] = <(-p_k), a_k>,  aa[k] = ||a_k||^2
//   rows [K,K+B): vv[b] = ||x_b||^2
// ---------------------------------------------------------------------------
__global__ __launch_bounds__(256) void hyp_stats_kernel(
    const float* __restrict__ inp, const float* __restrict__ p,
    const float* __restrict__ a,
    float* __restrict__ uu, float* __restrict__ ua,
    float* __restrict__ aa, float* __restrict__ vv) {
  const int wave = (blockIdx.x * blockDim.x + threadIdx.x) >> 5;
  const int lane = threadIdx.x & 31;
  if (wave < K_SZ) {
    const float* pr = p + (size_t)wave * D_SZ;
    const float* ar = a + (size_t)wave * D_SZ;
    float s_uu = 0.f, s_pa = 0.f, s_aa = 0.f;
#pragma unroll
    for (int i = 0; i < D_SZ / 128; ++i) {
      int d = i * 128 + lane * 4;
      float4 pv = *reinterpret_cast<const float4*>(pr + d);
      float4 av = *reinterpret_cast<const float4*>(ar + d);
      s_uu += pv.x * pv.x + pv.y * pv.y + pv.z * pv.z + pv.w * pv.w;
      s_pa += pv.x * av.x + pv.y * av.y + pv.z * av.z + pv.w * av.w;
      s_aa += av.x * av.x + av.y * av.y + av.z * av.z + av.w * av.w;
    }
#pragma unroll
    for (int off = 16; off > 0; off >>= 1) {
      s_uu += __shfl_xor(s_uu, off, 32);
      s_pa += __shfl_xor(s_pa, off, 32);
      s_aa += __shfl_xor(s_aa, off, 32);
    }
    if (lane == 0) { uu[wave] = s_uu; ua[wave] = -s_pa; aa[wave] = s_aa; }
  } else if (wave < K_SZ + B_SZ) {
    const int b = wave - K_SZ;
    const float* xr = inp + (size_t)b * D_SZ;
    float s = 0.f;
#pragma unroll
    for (int i = 0; i < D_SZ / 128; ++i) {
      int d = i * 128 + lane * 4;
      float4 xv = *reinterpret_cast<const float4*>(xr + d);
      s += xv.x * xv.x + xv.y * xv.y + xv.z * xv.z + xv.w * xv.w;
    }
#pragma unroll
    for (int off = 16; off > 0; off >>= 1) s += __shfl_xor(s, off, 32);
    if (lane == 0) vv[b] = s;
  }
}

// ---------------------------------------------------------------------------
// Pre-kernel 2: one-time f32 -> (bf16 hi, bf16 lo) split of inp/p/a into
// fragment-permuted planes.  x = hi + lo with |lo| <= ulp_bf16(x)/2, so
// A_hi*B_hi + A_hi*B_lo + A_lo*B_hi reconstructs ~fp32 products.
// One thread per (row, 32-col chunk).
// ---------------------------------------------------------------------------
__global__ __launch_bounds__(256) void hyp_split_kernel(
    const float* __restrict__ inp, const float* __restrict__ p,
    const float* __restrict__ a,
    __bf16* __restrict__ p_hi, __bf16* __restrict__ p_lo,
    __bf16* __restrict__ a_hi, __bf16* __restrict__ a_lo,
    __bf16* __restrict__ x_hi, __bf16* __restrict__ x_lo) {
  const int job = blockIdx.x * blockDim.x + threadIdx.x;
  if (job >= (2 * K_SZ + B_SZ) * NCHUNK) return;
  const int cid = job & (NCHUNK - 1);
  int rid = job >> 4;

  const float* src;
  __bf16 *dhi, *dlo;
  if (rid < K_SZ) {
    src = p + (size_t)rid * D_SZ;  dhi = p_hi + (size_t)rid * D_SZ;  dlo = p_lo + (size_t)rid * D_SZ;
  } else if (rid < 2 * K_SZ) {
    rid -= K_SZ;
    src = a + (size_t)rid * D_SZ;  dhi = a_hi + (size_t)rid * D_SZ;  dlo = a_lo + (size_t)rid * D_SZ;
  } else {
    rid -= 2 * K_SZ;
    src = inp + (size_t)rid * D_SZ; dhi = x_hi + (size_t)rid * D_SZ; dlo = x_lo + (size_t)rid * D_SZ;
  }
  src += cid * 32; dhi += cid * 32; dlo += cid * 32;

  float f[32];
#pragma unroll
  for (int i = 0; i < 8; ++i) {
    float4 t = reinterpret_cast<const float4*>(src)[i];
    f[i * 4 + 0] = t.x; f[i * 4 + 1] = t.y; f[i * 4 + 2] = t.z; f[i * 4 + 3] = t.w;
  }
  v8bf hv[4], lv[4];
#pragma unroll
  for (int g = 0; g < 4; ++g) {
#pragma unroll
    for (int e = 0; e < 8; ++e) {
      const int pos = g * 8 + e;
      const float x = f[frag_perm(pos)];
      const __bf16 h = (__bf16)x;
      hv[g][e] = h;
      lv[g][e] = (__bf16)(x - (float)h);
    }
  }
#pragma unroll
  for (int g = 0; g < 4; ++g) {
    reinterpret_cast<v8bf*>(dhi)[g] = hv[g];
    reinterpret_cast<v8bf*>(dlo)[g] = lv[g];
  }
}

// ---------------------------------------------------------------------------
// Shared epilogue: analytic hyperbolic-MLR from the two dot products.
// C/D layout: lane L -> col N = L&15, VGPR v -> row M = v + (L>>4)*8.
// ---------------------------------------------------------------------------
__device__ __forceinline__ void hyp_epilogue(
    const v8f& acc_pv, const v8f& acc_xa, int mtile, int ntile, int lane,
    const float* __restrict__ uu_v, const float* __restrict__ ua_v,
    const float* __restrict__ aa_v, const float* __restrict__ vv_v,
    float* __restrict__ out) {
  const int   r      = lane & 15;
  const int   k      = ntile * 16 + r;
  const float uu     = uu_v[k];
  const float ua     = ua_v[k];
  const float aa     = aa_v[k];
  const float a_norm = sqrtf(aa);
  const float sc     = sqrtf(CURV);
  const float lam    = 2.0f / (1.0f - CURV * uu);
  const float coef   = lam * a_norm / sc;
  const float beta   = 1.0f - CURV * uu;
#pragma unroll
  for (int v = 0; v < 8; ++v) {
    const int   b     = mtile * 16 + (lane >> 4) * 8 + v;
    const float vv    = vv_v[b];
    const float uv    = -acc_pv[v];
    const float xa    =  acc_xa[v];
    const float alpha = 1.0f + 2.0f * CURV * uv + CURV * vv;
    const float den   = 1.0f + 2.0f * CURV * uv + CURV * CURV * uu * vv;
    const float wa    = (alpha * ua + beta * xa) / den;
    const float ww    = (alpha * alpha * uu + 2.0f * alpha * beta * uv
                         + beta * beta * vv) / (den * den);
    const float num   = 2.0f * sc * wa;
    const float denom = a_norm * (1.0f - CURV * ww);
    out[(size_t)b * K_SZ + k] = coef * asinhf(num / denom);
  }
}

// ---------------------------------------------------------------------------
// FAST main kernel: fragments come pre-split/pre-permuted from the bf16
// planes, so each operand fragment is one contiguous 32-byte v16bf load.
// Inner loop = 6 vector loads + 6 v_wmma per K=32 chunk; near-zero VALU.
// ---------------------------------------------------------------------------
__global__ __launch_bounds__(128) void hyp_mlr_fast_kernel(
    const __bf16* __restrict__ x_hi, const __bf16* __restrict__ x_lo,
    const __bf16* __restrict__ p_hi, const __bf16* __restrict__ p_lo,
    const __bf16* __restrict__ a_hi, const __bf16* __restrict__ a_lo,
    const float* __restrict__ uu_v, const float* __restrict__ ua_v,
    const float* __restrict__ aa_v, const float* __restrict__ vv_v,
    float* __restrict__ out) {
  const int lane  = threadIdx.x & 31;
  const int wave  = threadIdx.x >> 5;
  const int tile  = blockIdx.x * 4 + wave;   // 1024 tiles total
  const int ntile = tile & 127;              // K / 16
  const int mtile = tile >> 7;               // B / 16
  const int r     = lane & 15;
  const int loff  = (lane >> 4) * 16;        // fragment offset within chunk

  const size_t arow = (size_t)(mtile * 16 + r) * D_SZ + loff;
  const size_t prow = (size_t)(ntile * 16 + r) * D_SZ + loff;

  const __bf16* xh = x_hi + arow; const __bf16* xl = x_lo + arow;
  const __bf16* ph = p_hi + prow; const __bf16* pl = p_lo + prow;
  const __bf16* qh = a_hi + prow; const __bf16* ql = a_lo + prow;

  v8f acc_pv = {};   // sum_d x[b,d] * p[k,d]   (uv = -acc_pv)
  v8f acc_xa = {};   // sum_d x[b,d] * a[k,d]

  for (int kk = 0; kk < D_SZ; kk += 32) {
    const v16bf ah = *reinterpret_cast<const v16bf*>(xh + kk);
    const v16bf al = *reinterpret_cast<const v16bf*>(xl + kk);
    const v16bf bh = *reinterpret_cast<const v16bf*>(ph + kk);
    const v16bf bl = *reinterpret_cast<const v16bf*>(pl + kk);
    const v16bf ch = *reinterpret_cast<const v16bf*>(qh + kk);
    const v16bf cl = *reinterpret_cast<const v16bf*>(ql + kk);
    acc_pv = __builtin_amdgcn_wmma_f32_16x16x32_bf16(false, ah, false, bh, (short)0, acc_pv, false, false);
    acc_pv = __builtin_amdgcn_wmma_f32_16x16x32_bf16(false, ah, false, bl, (short)0, acc_pv, false, false);
    acc_pv = __builtin_amdgcn_wmma_f32_16x16x32_bf16(false, al, false, bh, (short)0, acc_pv, false, false);
    acc_xa = __builtin_amdgcn_wmma_f32_16x16x32_bf16(false, ah, false, ch, (short)0, acc_xa, false, false);
    acc_xa = __builtin_amdgcn_wmma_f32_16x16x32_bf16(false, ah, false, cl, (short)0, acc_xa, false, false);
    acc_xa = __builtin_amdgcn_wmma_f32_16x16x32_bf16(false, al, false, ch, (short)0, acc_xa, false, false);
  }
  hyp_epilogue(acc_pv, acc_xa, mtile, ntile, lane, uu_v, ua_v, aa_v, vv_v, out);
}

// ---------------------------------------------------------------------------
// FALLBACK main kernel (workspace too small for planes): in-loop f32->bf16
// split, fragments built from f32 vector loads.  Same math, same layouts.
// ---------------------------------------------------------------------------
__device__ __forceinline__ void load_split_frag(const float* __restrict__ row,
                                                int col0, v16bf& hi, v16bf& lo) {
  const float4* q0 = reinterpret_cast<const float4*>(row + col0);
  const float4* q1 = reinterpret_cast<const float4*>(row + col0 + 16);
  float4 t0 = q0[0], t1 = q0[1], t2 = q1[0], t3 = q1[1];
  float f[16] = { t0.x, t0.y, t0.z, t0.w,  t1.x, t1.y, t1.z, t1.w,
                  t2.x, t2.y, t2.z, t2.w,  t3.x, t3.y, t3.z, t3.w };
#pragma unroll
  for (int e = 0; e < 16; ++e) {
    __bf16 h = (__bf16)f[e];
    hi[e] = h;
    lo[e] = (__bf16)(f[e] - (float)h);
  }
}

__global__ __launch_bounds__(128) void hyp_mlr_kernel(
    const float* __restrict__ inp, const float* __restrict__ p,
    const float* __restrict__ a,
    const float* __restrict__ uu_v, const float* __restrict__ ua_v,
    const float* __restrict__ aa_v, const float* __restrict__ vv_v,
    float* __restrict__ out) {
  const int lane  = threadIdx.x & 31;
  const int wave  = threadIdx.x >> 5;
  const int tile  = blockIdx.x * 4 + wave;
  const int ntile = tile & 127;
  const int mtile = tile >> 7;
  const int r     = lane & 15;
  const int kbase = (lane >> 4) << 3;

  const float* arow = inp + (size_t)(mtile * 16 + r) * D_SZ;
  const float* prow = p   + (size_t)(ntile * 16 + r) * D_SZ;
  const float* qrow = a   + (size_t)(ntile * 16 + r) * D_SZ;

  v8f acc_pv = {}; v8f acc_xa = {};
  for (int kk = 0; kk < D_SZ; kk += 32) {
    const int col0 = kk + kbase;
    v16bf ah, al, bh, bl, ch, cl;
    load_split_frag(arow, col0, ah, al);
    load_split_frag(prow, col0, bh, bl);
    load_split_frag(qrow, col0, ch, cl);
    acc_pv = __builtin_amdgcn_wmma_f32_16x16x32_bf16(false, ah, false, bh, (short)0, acc_pv, false, false);
    acc_pv = __builtin_amdgcn_wmma_f32_16x16x32_bf16(false, ah, false, bl, (short)0, acc_pv, false, false);
    acc_pv = __builtin_amdgcn_wmma_f32_16x16x32_bf16(false, al, false, bh, (short)0, acc_pv, false, false);
    acc_xa = __builtin_amdgcn_wmma_f32_16x16x32_bf16(false, ah, false, ch, (short)0, acc_xa, false, false);
    acc_xa = __builtin_amdgcn_wmma_f32_16x16x32_bf16(false, ah, false, cl, (short)0, acc_xa, false, false);
    acc_xa = __builtin_amdgcn_wmma_f32_16x16x32_bf16(false, al, false, ch, (short)0, acc_xa, false, false);
  }
  hyp_epilogue(acc_pv, acc_xa, mtile, ntile, lane, uu_v, ua_v, aa_v, vv_v, out);
}

extern "C" void kernel_launch(void* const* d_in, const int* in_sizes, int n_in,
                              void* d_out, int out_size, void* d_ws, size_t ws_size,
                              hipStream_t stream) {
  (void)in_sizes; (void)n_in; (void)out_size;
  const float* inp = (const float*)d_in[0];   // [B, D]
  const float* p   = (const float*)d_in[1];   // [K, D]
  const float* a   = (const float*)d_in[2];   // [K, D]
  float* out = (float*)d_out;                 // [B, K]

  // Workspace layout: stats (f32) then bf16 planes.
  char* ws = (char*)d_ws;
  const size_t stats_bytes = (size_t)(3 * K_SZ + B_SZ) * sizeof(float);  // 25088
  const size_t planeK = (size_t)K_SZ * D_SZ * sizeof(__bf16);            // 2 MB
  const size_t planeB = (size_t)B_SZ * D_SZ * sizeof(__bf16);            // 128 KB
  const size_t need   = stats_bytes + 4 * planeK + 2 * planeB;           // ~8.7 MB

  float* uu = (float*)ws;
  float* ua = uu + K_SZ;
  float* aa = ua + K_SZ;
  float* vv = aa + K_SZ;

  // Stats: one wave per row, (K + B) rows, 8 waves per 256-thread block.
  const int rows = K_SZ + B_SZ;               // 2176
  dim3 sblk(256), sgrd((rows * 32 + 255) / 256);
  hyp_stats_kernel<<<sgrd, sblk, 0, stream>>>(inp, p, a, uu, ua, aa, vv);

  if (ws_size >= need) {
    __bf16* p_hi = (__bf16*)(ws + stats_bytes);
    __bf16* p_lo = (__bf16*)((char*)p_hi + planeK);
    __bf16* a_hi = (__bf16*)((char*)p_lo + planeK);
    __bf16* a_lo = (__bf16*)((char*)a_hi + planeK);
    __bf16* x_hi = (__bf16*)((char*)a_lo + planeK);
    __bf16* x_lo = (__bf16*)((char*)x_hi + planeB);

    const int jobs = (2 * K_SZ + B_SZ) * NCHUNK;         // 67584
    hyp_split_kernel<<<dim3((jobs + 255) / 256), dim3(256), 0, stream>>>(
        inp, p, a, p_hi, p_lo, a_hi, a_lo, x_hi, x_lo);

    hyp_mlr_fast_kernel<<<dim3(256), dim3(128), 0, stream>>>(
        x_hi, x_lo, p_hi, p_lo, a_hi, a_lo, uu, ua, aa, vv, out);
  } else {
    hyp_mlr_kernel<<<dim3(256), dim3(128), 0, stream>>>(
        inp, p, a, uu, ua, aa, vv, out);
  }
}
